// GraphAttentionLayer_31610959299012
// MI455X (gfx1250) — matrix operand order repaired and checked by reading, not compiled, not run on hardware
//
#include <hip/hip_runtime.h>
#include <hip/hip_bf16.h>

typedef __attribute__((ext_vector_type(16))) _Float16 v16h;
typedef __attribute__((ext_vector_type(8)))  _Float16 v8h;
typedef __attribute__((ext_vector_type(8)))  float    v8f;

#define NN   4096
#define FF   64
#define BB   8
#define KSTEP 128
#define MTILE 32      // C-rows per workgroup in the big GEMM (2 x 16-row WMMA tiles)
#define APITCH 136    // LDS row pitch in halfs: 272B = 16B-aligned + 4-bank rotation

// ---------------------------------------------------------------------------
// Kernel 1: h[b,i,f] = inp[b,i,:] . W_w[f,:] + W_b[f];  h[:,0,:] = 0
// Writes f32 h (row-major [b][i][f]) and f16 transposed h16T ([b][f][j]).
// One wave handles (batch, 16-row tile); 2 K-steps x 4 N-tiles of WMMA.
// ---------------------------------------------------------------------------
__global__ void __launch_bounds__(256) k_linear(const float* __restrict__ inp,
                                                const float* __restrict__ Ww,
                                                const float* __restrict__ Wb,
                                                float* __restrict__ h32,
                                                _Float16* __restrict__ h16T) {
  const int lane = threadIdx.x & 31;
  const int wave = threadIdx.x >> 5;
  const int lin  = blockIdx.x * 8 + wave;      // 0..2047
  const int b    = lin >> 8;                   // batch
  const int i0   = (lin & 255) * 16;           // row tile
  const int half = lane >> 4, nl = lane & 15;

  // A fragments: rows of inp (M = nl), K pairs per documented 16-bit A layout
  v16h a0, a1;
  const float* arow = inp + ((size_t)b * NN + i0 + nl) * FF;
#pragma unroll
  for (int v = 0; v < 8; ++v) {
    int K0 = 2 * v + 8 * half + ((v >= 4) ? 8 : 0);
    a0[2 * v]     = (_Float16)arow[K0];
    a0[2 * v + 1] = (_Float16)arow[K0 + 1];
    a1[2 * v]     = (_Float16)arow[32 + K0];
    a1[2 * v + 1] = (_Float16)arow[32 + K0 + 1];
  }

#pragma unroll
  for (int t = 0; t < 4; ++t) {
    const int f0 = t * 16;
    // B[k,f] = Ww[f*64 + k]  (B is W transposed)
    v16h b0, b1;
    const float* wr = Ww + (size_t)(f0 + nl) * FF;
#pragma unroll
    for (int v = 0; v < 8; ++v) {
      int K0 = 2 * v + 8 * half + ((v >= 4) ? 8 : 0);
      b0[2 * v]     = (_Float16)wr[K0];
      b0[2 * v + 1] = (_Float16)wr[K0 + 1];
      b1[2 * v]     = (_Float16)wr[32 + K0];
      b1[2 * v + 1] = (_Float16)wr[32 + K0 + 1];
    }
    v8f c = {};
    c = __builtin_amdgcn_wmma_f32_16x16x32_f16(false, a0, false, b0, (short)0, c, false, false);
    c = __builtin_amdgcn_wmma_f32_16x16x32_f16(false, a1, false, b1, (short)0, c, false, false);

    const float bias = Wb[f0 + nl];
#pragma unroll
    for (int v = 0; v < 8; ++v) {
      const int i = i0 + v + 8 * half;       // C/D layout: M = v + 8*half
      const int f = f0 + nl;                 //             N = nl
      float val = c[v] + bias;
      if (i == 0) val = 0.0f;
      h32[((size_t)b * NN + i) * FF + f]     = val;
      h16T[((size_t)b * FF + f) * NN + i]    = (_Float16)val;
    }
  }
}

// ---------------------------------------------------------------------------
// Kernel 2 (dominant, ~17 GFLOP): per batch b,
//   C[i,f] = sum_{j=0..4094} (adj[j+1,i] > 0) * h[b,j,f]
// then fuse s_neigh[b,i] = sum_f C[i,f] * a[64+f, i].
// Workgroup = 32 C-rows (2 M-tiles); 8 waves = 8 batches sharing the binary
// A tile via LDS. Double-buffered adj staging through registers; vectorized
// b128 fragment loads (LDS + global).
// ---------------------------------------------------------------------------
__global__ void __launch_bounds__(256) k_gemm(const int* __restrict__ adj,
                                              const _Float16* __restrict__ h16T,
                                              const float* __restrict__ a_att,
                                              float* __restrict__ sneigh) {
  __shared__ __align__(16) _Float16 Alds[MTILE][APITCH];

  const int tid  = threadIdx.x;
  const int lane = tid & 31;
  const int b    = tid >> 5;                   // wave index = batch
  const int i0   = blockIdx.x * MTILE;
  const int half = lane >> 4, nl = lane & 15;
  const int sm   = tid & (MTILE - 1);          // staging: row
  const int skk  = tid >> 5;                   // staging: base column

  const _Float16* hb = h16T + (size_t)b * FF * NN;

  v8f c[2][4];
#pragma unroll
  for (int m2 = 0; m2 < 2; ++m2)
#pragma unroll
    for (int t = 0; t < 4; ++t) c[m2][t] = (v8f){};

  // Each thread owns 16 staged elements: (m = sm, kk = skk + 8*r)
  int areg[16];
#pragma unroll
  for (int r = 0; r < 16; ++r) {
    const int j = skk + 8 * r;                 // stage 0: k0 = 0
    areg[r] = adj[(size_t)(j + 1) * NN + i0 + sm];
  }

  for (int k0 = 0; k0 < NN - 1; k0 += KSTEP) {
    __syncthreads();
    // Store prefetched adj stage as binary f16 into LDS
#pragma unroll
    for (int r = 0; r < 16; ++r) {
      const int kk = skk + 8 * r;
      Alds[sm][kk] = (areg[r] > 0) ? (_Float16)1.0f : (_Float16)0.0f;
    }
    __syncthreads();

    // Kick off next stage's global loads; they complete under the WMMA work
    const int k0n = k0 + KSTEP;
    if (k0n < NN - 1) {
#pragma unroll
      for (int r = 0; r < 16; ++r) {
        const int j = k0n + skk + 8 * r;
        areg[r] = (j < NN - 1) ? adj[(size_t)(j + 1) * NN + i0 + sm] : 0;
      }
    }

#pragma unroll
    for (int kk0 = 0; kk0 < KSTEP; kk0 += 32) {
      // A fragments: two contiguous 16B runs per lane -> 2x ds_load_b128 each
      v16h af[2];
#pragma unroll
      for (int m2 = 0; m2 < 2; ++m2) {
        const _Float16* ar = &Alds[m2 * 16 + nl][kk0 + 16 * half];
        *((v8h*)&af[m2] + 0) = *(const v8h*)(ar);        // K = 16*half .. +7
        *((v8h*)&af[m2] + 1) = *(const v8h*)(ar + 8);    // K = 16*half+8 .. +15
      }
      const int jbase = k0 + kk0;
#pragma unroll
      for (int t = 0; t < 4; ++t) {
        // B fragment: one contiguous 32B run in j (h16T is [b][f][j])
        const v16h bf = *(const v16h*)(hb + (size_t)(t * 16 + nl) * NN + jbase + 16 * half);
        // reuse the B fragment across both M-tiles
#pragma unroll
        for (int m2 = 0; m2 < 2; ++m2)
          c[m2][t] = __builtin_amdgcn_wmma_f32_16x16x32_f16(false, af[m2], false, bf,
                                                            (short)0, c[m2][t], false, false);
      }
    }
  }

  // Fused epilogue: s_neigh[b,i] = sum_f C[i,f] * a[64+f, i]
#pragma unroll
  for (int m2 = 0; m2 < 2; ++m2) {
#pragma unroll
    for (int v = 0; v < 8; ++v) {
      const int i = i0 + m2 * 16 + v + 8 * half;
      float p = 0.0f;
#pragma unroll
      for (int t = 0; t < 4; ++t) {
        const float a2 = a_att[(size_t)(FF + t * 16 + nl) * NN + i];
        p += c[m2][t][v] * a2;
      }
      // reduce across the 16 lanes of this lane-half (N dimension)
#pragma unroll
      for (int off = 1; off < 16; off <<= 1)
        p += __shfl_xor(p, off, 32);
      if (nl == 0)
        sneigh[(size_t)b * NN + i] = p;
    }
  }
}

// ---------------------------------------------------------------------------
// Kernel 3: hsum[b,f] = sum_i h[b,i,f]
// ---------------------------------------------------------------------------
__global__ void k_hsum(const float* __restrict__ h32, float* __restrict__ hsum) {
  const int b = blockIdx.x;
  const int f = threadIdx.x;       // 64 threads
  const float* p = h32 + (size_t)b * NN * FF + f;
  float s = 0.0f;
  for (int i = 0; i < NN; ++i) s += p[(size_t)i * FF];
  hsum[b * FF + f] = s;
}

// ---------------------------------------------------------------------------
// Kernel 4: s1[b,i] = sum_f h[b,i,f] * a[f,i]
// ---------------------------------------------------------------------------
__global__ void k_s1(const float* __restrict__ h32, const float* __restrict__ a_att,
                     float* __restrict__ s1) {
  const int idx = blockIdx.x * 256 + threadIdx.x;   // BB*NN = 32768
  if (idx >= BB * NN) return;
  const int i = idx & (NN - 1);
  const float* hr = h32 + (size_t)idx * FF;
  float s = 0.0f;
#pragma unroll 8
  for (int f = 0; f < FF; ++f) s += hr[f] * a_att[(size_t)f * NN + i];
  s1[idx] = s;
}

// ---------------------------------------------------------------------------
// Kernel 5: out[b,i,f] = leaky( s[b,i] * hsum[b,f] ), s = s1 + sneigh, s[:,0]=0
// ---------------------------------------------------------------------------
__global__ void k_out(const float* __restrict__ s1, const float* __restrict__ sneigh,
                      const float* __restrict__ hsum, float* __restrict__ out) {
  const size_t idx = (size_t)blockIdx.x * 256 + threadIdx.x;  // 2,097,152
  const int f  = (int)(idx & (FF - 1));
  const size_t bi = idx >> 6;
  const int i  = (int)(bi & (NN - 1));
  const int b  = (int)(bi >> 12);
  const float s = (i == 0) ? 0.0f : (s1[bi] + sneigh[bi]);
  const float v = s * hsum[b * FF + f];
  out[idx] = (v >= 0.0f) ? v : 0.01f * v;
}

// ---------------------------------------------------------------------------
extern "C" void kernel_launch(void* const* d_in, const int* in_sizes, int n_in,
                              void* d_out, int out_size, void* d_ws, size_t ws_size,
                              hipStream_t stream) {
  const float* inp  = (const float*)d_in[0];   // [8,4096,64]
  const int*   adj  = (const int*)  d_in[1];   // [4096,4096]
  const float* Ww   = (const float*)d_in[2];   // [64,64]
  const float* Wb   = (const float*)d_in[3];   // [64]
  const float* aatt = (const float*)d_in[4];   // [128,4096]
  float* out = (float*)d_out;                  // [8,4096,64]

  char* ws = (char*)d_ws;
  float*    h32    = (float*)   (ws);                            // 8 MB
  _Float16* h16T   = (_Float16*)(ws + 8u * 1024 * 1024);         // 4 MB
  float*    sneigh = (float*)   (ws + 12u * 1024 * 1024);        // 128 KB
  float*    s1     = (float*)   (ws + 12u * 1024 * 1024 + 131072);
  float*    hsum   = (float*)   (ws + 12u * 1024 * 1024 + 262144);

  k_linear<<<256, 256, 0, stream>>>(inp, Ww, Wb, h32, h16T);
  k_gemm  <<<NN / MTILE, 256, 0, stream>>>(adj, h16T, aatt, sneigh);
  k_hsum  <<<BB, FF, 0, stream>>>(h32, hsum);
  k_s1    <<<(BB * NN) / 256, 256, 0, stream>>>(h32, aatt, s1);
  k_out   <<<(BB * NN * FF) / 256, 256, 0, stream>>>(s1, sneigh, hsum, out);
}
